// Net_81922206204539
// MI455X (gfx1250) — compile-verified, matrix operand order checked by reading, not run on hardware
//
#include <hip/hip_runtime.h>
#include <hip/hip_bf16.h>
#include <math.h>

// ---------------------------------------------------------------------------
// PointNet-ish pipeline for MI455X (gfx1250, wave32, WMMA 16x16x32 f16->f32).
// All GEMMs go through v_wmma_f32_16x16x32_f16; BN/bias/ReLU/max fused into
// epilogues. Round 2: BN scale/shift hoisted out of the hot loops into
// registers, and the reused weight matrices (c3: 256KB, lin1: 128KB) staged
// into CDNA5's 320KB-per-WG LDS so the streaming GEMM loops hit ds_load_b128
// instead of re-reading L2 every iteration.
// ---------------------------------------------------------------------------

#define NB   16
#define NF   16
#define NPP  256
#define NKNN 4
#define NPTS (NB*NF*NPP)     // 65536
#define NROWS (NPTS*NKNN)    // 262144
#define NOUT 40
#define BN_EPS 1e-5f

typedef __attribute__((ext_vector_type(16))) _Float16 v16h;
typedef __attribute__((ext_vector_type(8)))  _Float16 v8h;
typedef __attribute__((ext_vector_type(8)))  float    v8f;

__device__ __forceinline__ v8f wmma32(v16h a, v16h b, v8f c) {
  return __builtin_amdgcn_wmma_f32_16x16x32_f16(false, a, false, b, (short)0, c,
                                                false, false);
}

// 16x32 fragment load from a row-major f16 buffer (works for A rows and for
// B == W[cout][cin] since B[k][n] = W[n][k]).  Per CDNA5 ISA layout:
// lanes 0-15: row=lane, K = [k0..k0+7] and [k0+16..k0+23]
// lanes16-31: row=lane-16, K = [k0+8..k0+15] and [k0+24..k0+31]
__device__ __forceinline__ v16h ld_frag(const _Float16* p, int ld, int lane, int k0) {
  int r  = lane & 15;
  int kb = k0 + ((lane >> 4) << 3);
  const _Float16* q = p + r * ld + kb;
  v16h out;
  *(v8h*)&out         = *(const v8h*)q;
  *(((v8h*)&out) + 1) = *(const v8h*)(q + 16);
  return out;
}

// Fold BN (and optionally the linear bias) into per-column scale/shift.
// STN style  relu(bn(y+b))  : z = max(s*y + t_folded, 0)  (bias folded into t)
// MLP style  bn(relu(y+b))  : z = s*max(y + b, 0) + t      (bias kept separate)
__device__ __forceinline__ void bn_fold(int col, const float* g, const float* bb,
                                        const float* rm, const float* rv,
                                        float& s, float& t) {
  s = g[col] * rsqrtf(rv[col] + BN_EPS);
  t = bb[col] - rm[col] * s;
}

// Fallback epilogues for the one-shot (non-looped) kernels.
__device__ __forceinline__ float ep_stn(float y, int col, const float* b,
    const float* g, const float* bb, const float* rm, const float* rv) {
  float s, t;
  bn_fold(col, g, bb, rm, rv, s, t);
  return fmaxf(s * (y + b[col]) + t, 0.f);
}
__device__ __forceinline__ float ep_mlp(float y, int col, const float* b,
    const float* g, const float* bb, const float* rm, const float* rv) {
  float s, t;
  bn_fold(col, g, bb, rm, rv, s, t);
  return s * fmaxf(y + b[col], 0.f) + t;
}

// Monotonic float<->uint key for atomic min/max on floats.
__device__ __forceinline__ unsigned fkey(float f) {
  unsigned u = __float_as_uint(f);
  return (u & 0x80000000u) ? ~u : (u | 0x80000000u);
}
__device__ __forceinline__ float funkey(unsigned k) {
  unsigned u = (k & 0x80000000u) ? (k & 0x7fffffffu) : ~k;
  return __uint_as_float(u);
}

// --------------------------- weight conversion -----------------------------
__global__ void k_convert(_Float16* __restrict__ dst, const float* __restrict__ src,
                          int cout, int cin, int cinp) {
  int i = blockIdx.x * 256 + threadIdx.x;
  int tot = cout * cinp;
  if (i >= tot) return;
  int r = i / cinp, c = i % cinp;
  dst[i] = (c < cin) ? (_Float16)src[r * cin + c] : (_Float16)0.f;
}

// ------------------------- STN point MLP + max -----------------------------
// One workgroup per batch; streams 4096 points in 16-row tiles through
// 3->64->128->1024 with fused BN+ReLU, max-reducing into LDS.  W3 (256KB)
// lives in LDS for the whole kernel.
__global__ __launch_bounds__(256) void k_stn_point(
    const float* __restrict__ pos,
    const _Float16* __restrict__ W1, const float* b1, const float* g1, const float* bb1, const float* rm1, const float* rv1,
    const _Float16* __restrict__ W2, const float* b2, const float* g2, const float* bb2, const float* rm2, const float* rv2,
    const _Float16* __restrict__ W3, const float* b3, const float* g3, const float* bb3, const float* rm3, const float* rv3,
    float* __restrict__ stnfeat) {
  __shared__ __align__(16) _Float16 sW3[1024 * 128];   // 256 KB staged weights
  __shared__ __align__(16) _Float16 sA[16 * 32];
  __shared__ __align__(16) _Float16 sH1[16 * 64];
  __shared__ __align__(16) _Float16 sH2[16 * 128];
  __shared__ float sAcc[1024];
  int tid = threadIdx.x, lane = tid & 31, wv = tid >> 5;
  int b = blockIdx.x;
  // Stage c3 weights into LDS (one-time 256KB pull through L2).
  for (int i = tid; i < 1024 * 128 / 8; i += 256)
    ((v8h*)sW3)[i] = ((const v8h*)W3)[i];
  for (int i = tid; i < 1024; i += 256) sAcc[i] = -INFINITY;
  for (int i = tid; i < 16 * 32; i += 256) sA[i] = (_Float16)0.f;
  // Hoist folded BN params; each wave's output columns are loop-invariant.
  int colw = (lane & 15);
  float s1 = 0.f, t1 = 0.f;
  if (wv < 4) {
    int col = wv * 16 + colw;
    bn_fold(col, g1, bb1, rm1, rv1, s1, t1);
    t1 += s1 * b1[col];
  }
  float s2, t2;
  {
    int col = wv * 16 + colw;
    bn_fold(col, g2, bb2, rm2, rv2, s2, t2);
    t2 += s2 * b2[col];
  }
  float s3[8], t3[8];
  for (int nt = 0; nt < 8; ++nt) {
    int col = (wv * 8 + nt) * 16 + colw;
    bn_fold(col, g3, bb3, rm3, rv3, s3[nt], t3[nt]);
    t3[nt] += s3[nt] * b3[col];
  }
  __syncthreads();
  for (int it = 0; it < 256; ++it) {
    int row0 = b * 4096 + it * 16;
    if (tid < 48) {
      int r = tid / 3, c = tid % 3;
      sA[r * 32 + c] = (_Float16)pos[(row0 + r) * 3 + c];
    }
    __syncthreads();
    if (wv < 4) {  // c1: 3(pad32) -> 64
      v8f c = {};
      c = wmma32(ld_frag(sA, 32, lane, 0), ld_frag(W1 + wv * 16 * 32, 32, lane, 0), c);
      int col = wv * 16 + colw, mh = (lane >> 4) << 3;
      for (int r = 0; r < 8; ++r)
        sH1[(mh + r) * 64 + col] = (_Float16)fmaxf(s1 * c[r] + t1, 0.f);
    }
    __syncthreads();
    {  // c2: 64 -> 128
      v8f c = {};
      v16h a0 = ld_frag(sH1, 64, lane, 0), a1 = ld_frag(sH1, 64, lane, 32);
      const _Float16* Wt = W2 + wv * 16 * 64;
      c = wmma32(a0, ld_frag(Wt, 64, lane, 0), c);
      c = wmma32(a1, ld_frag(Wt, 64, lane, 32), c);
      int col = wv * 16 + colw, mh = (lane >> 4) << 3;
      for (int r = 0; r < 8; ++r)
        sH2[(mh + r) * 128 + col] = (_Float16)fmaxf(s2 * c[r] + t2, 0.f);
    }
    __syncthreads();
    {  // c3: 128 -> 1024 (weights from LDS), fused max over rows
      v16h a0 = ld_frag(sH2, 128, lane, 0),  a1 = ld_frag(sH2, 128, lane, 32);
      v16h a2 = ld_frag(sH2, 128, lane, 64), a3 = ld_frag(sH2, 128, lane, 96);
      for (int nt = 0; nt < 8; ++nt) {
        int ntile = wv * 8 + nt;
        const _Float16* Wt = sW3 + ntile * 16 * 128;
        v8f c = {};
        c = wmma32(a0, ld_frag(Wt, 128, lane, 0), c);
        c = wmma32(a1, ld_frag(Wt, 128, lane, 32), c);
        c = wmma32(a2, ld_frag(Wt, 128, lane, 64), c);
        c = wmma32(a3, ld_frag(Wt, 128, lane, 96), c);
        float m = -INFINITY;
        for (int r = 0; r < 8; ++r)
          m = fmaxf(m, fmaxf(s3[nt] * c[r] + t3[nt], 0.f));
        m = fmaxf(m, __shfl_xor(m, 16, 32));
        if (lane < 16) sAcc[ntile * 16 + colw] = fmaxf(sAcc[ntile * 16 + colw], m);
      }
    }
    __syncthreads();
  }
  for (int i = tid; i < 1024; i += 256) stnfeat[b * 1024 + i] = sAcc[i];
}

// ------------------------- STN FC head (16 rows) ---------------------------
__global__ __launch_bounds__(256) void k_stn_fc(
    const float* __restrict__ stnfeat,
    const _Float16* __restrict__ Wf1, const float* b4, const float* g4, const float* bb4, const float* rm4, const float* rv4,
    const _Float16* __restrict__ Wf2, const float* b5, const float* g5, const float* bb5, const float* rm5, const float* rv5,
    const float* __restrict__ Wf3, const float* __restrict__ bf3,
    float* __restrict__ trans) {
  __shared__ __align__(16) _Float16 sX[16 * 1024];
  __shared__ __align__(16) _Float16 sH[16 * 512];
  __shared__ __align__(16) _Float16 sG[16 * 256];
  int tid = threadIdx.x, lane = tid & 31, wv = tid >> 5;
  for (int i = tid; i < 16 * 1024; i += 256) sX[i] = (_Float16)stnfeat[i];
  __syncthreads();
  for (int nt = 0; nt < 4; ++nt) {  // fc1: 1024 -> 512
    int ntile = wv * 4 + nt;
    const _Float16* Wt = Wf1 + ntile * 16 * 1024;
    v8f c = {};
    for (int ks = 0; ks < 32; ++ks)
      c = wmma32(ld_frag(sX, 1024, lane, ks * 32), ld_frag(Wt, 1024, lane, ks * 32), c);
    int col = ntile * 16 + (lane & 15), mh = (lane >> 4) << 3;
    for (int r = 0; r < 8; ++r)
      sH[(mh + r) * 512 + col] = (_Float16)ep_stn(c[r], col, b4, g4, bb4, rm4, rv4);
  }
  __syncthreads();
  for (int nt = 0; nt < 2; ++nt) {  // fc2: 512 -> 256
    int ntile = wv * 2 + nt;
    const _Float16* Wt = Wf2 + ntile * 16 * 512;
    v8f c = {};
    for (int ks = 0; ks < 16; ++ks)
      c = wmma32(ld_frag(sH, 512, lane, ks * 32), ld_frag(Wt, 512, lane, ks * 32), c);
    int col = ntile * 16 + (lane & 15), mh = (lane >> 4) << 3;
    for (int r = 0; r < 8; ++r)
      sG[(mh + r) * 256 + col] = (_Float16)ep_stn(c[r], col, b5, g5, bb5, rm5, rv5);
  }
  __syncthreads();
  if (tid < 144) {  // fc3: 256 -> 9, + eye(3)
    int r = tid / 9, o = tid % 9;
    float acc = bf3[o] + ((o == 0 || o == 4 || o == 8) ? 1.f : 0.f);
    for (int j = 0; j < 256; ++j) acc += (float)sG[r * 256 + j] * Wf3[o * 256 + j];
    trans[r * 9 + o] = acc;
  }
}

// ----------------------- apply 3x3 transform -------------------------------
__global__ void k_transform(const float* __restrict__ pos, const float* __restrict__ trans,
                            float* __restrict__ pos_t) {
  int n = blockIdx.x * 256 + threadIdx.x;
  if (n >= NPTS) return;
  int b = n >> 12;
  const float* T = trans + b * 9;
  float x = pos[n * 3], y = pos[n * 3 + 1], z = pos[n * 3 + 2];
  pos_t[n * 3 + 0] = x * T[0] + y * T[3] + z * T[6];
  pos_t[n * 3 + 1] = x * T[1] + y * T[4] + z * T[7];
  pos_t[n * 3 + 2] = x * T[2] + y * T[5] + z * T[8];
}

// ----------------------- per-column min/max of pos_t -----------------------
__global__ void k_mm_init(unsigned* mm) {
  int i = threadIdx.x;
  if (i < 3) mm[i] = 0xFFFFFFFFu;
  else if (i < 6) mm[i] = 0u;
}
__global__ void k_minmax(const float* __restrict__ pos_t, unsigned* mm) {
  int t = blockIdx.x * 256 + threadIdx.x;
  unsigned lmin[3] = {~0u, ~0u, ~0u}, lmax[3] = {0u, 0u, 0u};
  for (int n = t; n < NPTS; n += gridDim.x * 256)
    for (int c = 0; c < 3; ++c) {
      unsigned k = fkey(pos_t[n * 3 + c]);
      lmin[c] = min(lmin[c], k);
      lmax[c] = max(lmax[c], k);
    }
  for (int c = 0; c < 3; ++c) {
    atomicMin(&mm[c], lmin[c]);
    atomicMax(&mm[3 + c], lmax[c]);
  }
}

// -------------------- KNN vs previous frame (T=1, K=4) ---------------------
__global__ __launch_bounds__(256) void k_knn(const float* __restrict__ pos_t,
                                             const unsigned* __restrict__ mm,
                                             int* __restrict__ nbr) {
  __shared__ float sC[NPP * 4];
  int bf = blockIdx.x, b = bf >> 4, f = bf & 15;
  int cf = (f == 0) ? 0 : f - 1;
  int tid = threadIdx.x;
  float mn[3], inv[3];
  for (int c = 0; c < 3; ++c) {
    float lo = funkey(mm[c]), hi = funkey(mm[3 + c]);
    mn[c] = lo;
    inv[c] = 1.f / (hi - lo);
  }
  // seq column: (f+1-1)/15 * STF*sqrt(3)
  const float wsc = 0.01f * 1.7320508075688772f / 15.f;
  int cbase = (b * 16 + cf) * NPP;
  {
    int cp = cbase + tid;
    sC[tid * 4 + 0] = (pos_t[cp * 3 + 0] - mn[0]) * inv[0];
    sC[tid * 4 + 1] = (pos_t[cp * 3 + 1] - mn[1]) * inv[1];
    sC[tid * 4 + 2] = (pos_t[cp * 3 + 2] - mn[2]) * inv[2];
    sC[tid * 4 + 3] = (float)cf * wsc;
  }
  __syncthreads();
  int n = (b * 16 + f) * NPP + tid;
  float p0 = (pos_t[n * 3 + 0] - mn[0]) * inv[0];
  float p1 = (pos_t[n * 3 + 1] - mn[1]) * inv[1];
  float p2 = (pos_t[n * 3 + 2] - mn[2]) * inv[2];
  float p3 = (float)f * wsc;
  float bd[4] = {INFINITY, INFINITY, INFINITY, INFINITY};
  int bi[4] = {0, 0, 0, 0};
  for (int j = 0; j < NPP; ++j) {
    float d0 = p0 - sC[j * 4], d1 = p1 - sC[j * 4 + 1];
    float d2 = p2 - sC[j * 4 + 2], d3 = p3 - sC[j * 4 + 3];
    float d = d0 * d0 + d1 * d1 + d2 * d2 + d3 * d3;
    if (d < bd[3]) {
      int k = 3;
      while (k > 0 && bd[k - 1] > d) { bd[k] = bd[k - 1]; bi[k] = bi[k - 1]; --k; }
      bd[k] = d; bi[k] = j;
    }
  }
  for (int k = 0; k < NKNN; ++k) nbr[n * NKNN + k] = cbase + bi[k];
}

// ------------- edge conv: build msg + first MLP block (6->64) --------------
__global__ __launch_bounds__(256) void k_edge0(
    const float* __restrict__ pos_t, const int* __restrict__ nbr,
    const _Float16* __restrict__ W, const float* b, const float* g,
    const float* bb, const float* rm, const float* rv,
    _Float16* __restrict__ out) {
  __shared__ __align__(16) _Float16 sA[32 * 32];
  int tid = threadIdx.x, lane = tid & 31, wv = tid >> 5;
  for (int i = tid; i < 32 * 32; i += 256) sA[i] = (_Float16)0.f;
  __syncthreads();
  if (tid < 32) {
    int R = blockIdx.x * 32 + tid;
    int n = R >> 2;
    int j = nbr[R];
    float x0 = pos_t[n * 3], x1 = pos_t[n * 3 + 1], x2 = pos_t[n * 3 + 2];
    sA[tid * 32 + 0] = (_Float16)x0;
    sA[tid * 32 + 1] = (_Float16)x1;
    sA[tid * 32 + 2] = (_Float16)x2;
    sA[tid * 32 + 3] = (_Float16)(pos_t[j * 3] - x0);
    sA[tid * 32 + 4] = (_Float16)(pos_t[j * 3 + 1] - x1);
    sA[tid * 32 + 5] = (_Float16)(pos_t[j * 3 + 2] - x2);
  }
  __syncthreads();
  int rowtile = wv >> 2, ntile = wv & 3;
  v8f c = {};
  c = wmma32(ld_frag(sA + rowtile * 16 * 32, 32, lane, 0),
             ld_frag(W + ntile * 16 * 32, 32, lane, 0), c);
  int col = ntile * 16 + (lane & 15), mh = (lane >> 4) << 3;
  size_t base = (size_t)blockIdx.x * 32;
  for (int r = 0; r < 8; ++r)
    out[(base + rowtile * 16 + mh + r) * 64 + col] =
        (_Float16)ep_mlp(c[r], col, b, g, bb, rm, rv);
}

// ----------------- generic 64->64 GEMM over NROWS rows ---------------------
// MODE 0: bias only (q/k/v).  MODE 1: mlp-block epilogue bn(relu(y+b)).
template <int MODE>
__global__ __launch_bounds__(256) void k_gemm64(
    const _Float16* __restrict__ in, const _Float16* __restrict__ W,
    const float* b, const float* g, const float* bb, const float* rm, const float* rv,
    _Float16* __restrict__ out) {
  __shared__ __align__(16) _Float16 sA[32 * 64];
  int tid = threadIdx.x, lane = tid & 31, wv = tid >> 5;
  size_t base = (size_t)blockIdx.x * 32 * 64;
  ((v8h*)sA)[tid] = ((const v8h*)(in + base))[tid];
  __syncthreads();
  int rowtile = wv >> 2, ntile = wv & 3;
  const _Float16* Ap = sA + rowtile * 16 * 64;
  const _Float16* Wt = W + ntile * 16 * 64;
  v8f c = {};
  c = wmma32(ld_frag(Ap, 64, lane, 0), ld_frag(Wt, 64, lane, 0), c);
  c = wmma32(ld_frag(Ap, 64, lane, 32), ld_frag(Wt, 64, lane, 32), c);
  int col = ntile * 16 + (lane & 15), mh = (lane >> 4) << 3;
  for (int r = 0; r < 8; ++r) {
    float y = (MODE == 1) ? ep_mlp(c[r], col, b, g, bb, rm, rv) : (c[r] + b[col]);
    out[base + (size_t)(rowtile * 16 + mh + r) * 64 + col] = (_Float16)y;
  }
}

// ----------------- per-point MHA (K=4 seq, 8 heads, dh=8) ------------------
__global__ __launch_bounds__(256) void k_attn(
    const _Float16* __restrict__ Q, const _Float16* __restrict__ Km,
    const _Float16* __restrict__ V, _Float16* __restrict__ out) {
  int t = blockIdx.x * 256 + threadIdx.x;
  int n = t >> 3, h = t & 7;
  float q[4][8], k[4][8], v[4][8];
  for (int kk = 0; kk < 4; ++kk) {
    size_t base = (size_t)(n * 4 + kk) * 64 + h * 8;
    v8h qv = *(const v8h*)(Q + base);
    v8h kv = *(const v8h*)(Km + base);
    v8h vv = *(const v8h*)(V + base);
    for (int d = 0; d < 8; ++d) {
      q[kk][d] = (float)qv[d]; k[kk][d] = (float)kv[d]; v[kk][d] = (float)vv[d];
    }
  }
  const float scale = 0.35355339059327373f;  // 1/sqrt(8)
  for (int qi = 0; qi < 4; ++qi) {
    float s[4], m = -INFINITY;
    for (int ki = 0; ki < 4; ++ki) {
      float d = 0.f;
      for (int dd = 0; dd < 8; ++dd) d += q[qi][dd] * k[ki][dd];
      s[ki] = d * scale;
      m = fmaxf(m, s[ki]);
    }
    float se = 0.f;
    for (int ki = 0; ki < 4; ++ki) { s[ki] = __expf(s[ki] - m); se += s[ki]; }
    float inv = 1.f / se;
    size_t base = (size_t)(n * 4 + qi) * 64 + h * 8;
    v8h ov;
    for (int dd = 0; dd < 8; ++dd) {
      float a = 0.f;
      for (int ki = 0; ki < 4; ++ki) a += s[ki] * v[ki][dd];
      ov[dd] = (_Float16)(a * inv);
    }
    *(v8h*)(out + base) = ov;
  }
}

// ------------- o-projection + max over K=4 neighbors -> x1 -----------------
__global__ __launch_bounds__(256) void k_oproj_max(
    const _Float16* __restrict__ in, const _Float16* __restrict__ W,
    const float* __restrict__ bias, _Float16* __restrict__ x1) {
  __shared__ __align__(16) _Float16 sA[32 * 64];
  int tid = threadIdx.x, lane = tid & 31, wv = tid >> 5;
  size_t base = (size_t)blockIdx.x * 32 * 64;
  ((v8h*)sA)[tid] = ((const v8h*)(in + base))[tid];
  __syncthreads();
  int rowtile = wv >> 2, ntile = wv & 3;
  const _Float16* Ap = sA + rowtile * 16 * 64;
  const _Float16* Wt = W + ntile * 16 * 64;
  v8f c = {};
  c = wmma32(ld_frag(Ap, 64, lane, 0), ld_frag(Wt, 64, lane, 0), c);
  c = wmma32(ld_frag(Ap, 64, lane, 32), ld_frag(Wt, 64, lane, 32), c);
  int col = ntile * 16 + (lane & 15);
  float bc = bias[col];
  float p0 = fmaxf(fmaxf(c[0], c[1]), fmaxf(c[2], c[3])) + bc;  // rows M..M+3
  float p1 = fmaxf(fmaxf(c[4], c[5]), fmaxf(c[6], c[7])) + bc;  // rows M+4..M+7
  int half = lane >> 4;
  int n0 = blockIdx.x * 8 + rowtile * 4 + half * 2;
  x1[(size_t)n0 * 64 + col] = (_Float16)p0;
  x1[(size_t)(n0 + 1) * 64 + col] = (_Float16)p1;
}

// ------------- lin1 (64->1024) + per-batch max over 4096 rows --------------
// One workgroup per batch; lin1 weights (128KB) staged in LDS, BN hoisted.
__global__ __launch_bounds__(256) void k_lin1max(
    const _Float16* __restrict__ x1, const _Float16* __restrict__ W,
    const float* bL, const float* gL, const float* bbL, const float* rmL, const float* rvL,
    float* __restrict__ hfeat) {
  __shared__ __align__(16) _Float16 sW[1024 * 64];    // 128 KB staged weights
  __shared__ float sAcc[1024];
  int tid = threadIdx.x, lane = tid & 31, wv = tid >> 5, b = blockIdx.x;
  for (int i = tid; i < 1024 * 64 / 8; i += 256)
    ((v8h*)sW)[i] = ((const v8h*)W)[i];
  for (int i = tid; i < 1024; i += 256) sAcc[i] = -INFINITY;
  int colw = lane & 15;
  float sv[8], tv[8], bv[8];
  for (int nt = 0; nt < 8; ++nt) {
    int col = (wv * 8 + nt) * 16 + colw;
    bn_fold(col, gL, bbL, rmL, rvL, sv[nt], tv[nt]);
    bv[nt] = bL[col];
  }
  __syncthreads();
  for (int it = 0; it < 256; ++it) {
    const _Float16* arow = x1 + (size_t)(b * 4096 + it * 16) * 64;
    v16h a0 = ld_frag(arow, 64, lane, 0);
    v16h a1 = ld_frag(arow, 64, lane, 32);
    for (int nt = 0; nt < 8; ++nt) {
      int ntile = wv * 8 + nt;
      const _Float16* Wt = sW + ntile * 16 * 64;
      v8f c = {};
      c = wmma32(a0, ld_frag(Wt, 64, lane, 0), c);
      c = wmma32(a1, ld_frag(Wt, 64, lane, 32), c);
      float m = -INFINITY;
      for (int r = 0; r < 8; ++r)
        m = fmaxf(m, sv[nt] * fmaxf(c[r] + bv[nt], 0.f) + tv[nt]);
      m = fmaxf(m, __shfl_xor(m, 16, 32));
      if (lane < 16) sAcc[ntile * 16 + colw] = fmaxf(sAcc[ntile * 16 + colw], m);
    }
  }
  __syncthreads();
  for (int i = tid; i < 1024; i += 256) hfeat[b * 1024 + i] = sAcc[i];
}

// ------------------ classifier head + log_softmax (16 rows) ----------------
__global__ __launch_bounds__(256) void k_head(
    const float* __restrict__ hfeat,
    const _Float16* __restrict__ Wh1, const float* b1, const float* g1, const float* bb1, const float* rm1, const float* rv1,
    const _Float16* __restrict__ Wh2, const float* b2, const float* g2, const float* bb2, const float* rm2, const float* rv2,
    const float* __restrict__ Wo, const float* __restrict__ bo,
    float* __restrict__ out) {
  __shared__ __align__(16) _Float16 sX[16 * 1024];
  __shared__ __align__(16) _Float16 sH[16 * 512];
  __shared__ __align__(16) _Float16 sG[16 * 256];
  __shared__ float sO[16 * NOUT];
  int tid = threadIdx.x, lane = tid & 31, wv = tid >> 5;
  for (int i = tid; i < 16 * 1024; i += 256) sX[i] = (_Float16)hfeat[i];
  __syncthreads();
  for (int nt = 0; nt < 4; ++nt) {  // head1: 1024 -> 512
    int ntile = wv * 4 + nt;
    const _Float16* Wt = Wh1 + ntile * 16 * 1024;
    v8f c = {};
    for (int ks = 0; ks < 32; ++ks)
      c = wmma32(ld_frag(sX, 1024, lane, ks * 32), ld_frag(Wt, 1024, lane, ks * 32), c);
    int col = ntile * 16 + (lane & 15), mh = (lane >> 4) << 3;
    for (int r = 0; r < 8; ++r)
      sH[(mh + r) * 512 + col] = (_Float16)ep_mlp(c[r], col, b1, g1, bb1, rm1, rv1);
  }
  __syncthreads();
  for (int nt = 0; nt < 2; ++nt) {  // head2: 512 -> 256
    int ntile = wv * 2 + nt;
    const _Float16* Wt = Wh2 + ntile * 16 * 512;
    v8f c = {};
    for (int ks = 0; ks < 16; ++ks)
      c = wmma32(ld_frag(sH, 512, lane, ks * 32), ld_frag(Wt, 512, lane, ks * 32), c);
    int col = ntile * 16 + (lane & 15), mh = (lane >> 4) << 3;
    for (int r = 0; r < 8; ++r)
      sG[(mh + r) * 256 + col] = (_Float16)ep_mlp(c[r], col, b2, g2, bb2, rm2, rv2);
  }
  __syncthreads();
  for (int i = tid; i < 16 * NOUT; i += 256) {  // out: 256 -> 40
    int r = i / NOUT, o = i % NOUT;
    float acc = bo[o];
    for (int j = 0; j < 256; ++j) acc += (float)sG[r * 256 + j] * Wo[o * 256 + j];
    sO[i] = acc;
  }
  __syncthreads();
  if (tid < 16) {  // log_softmax per row
    float m = -INFINITY;
    for (int o = 0; o < NOUT; ++o) m = fmaxf(m, sO[tid * NOUT + o]);
    float se = 0.f;
    for (int o = 0; o < NOUT; ++o) se += __expf(sO[tid * NOUT + o] - m);
    float ls = logf(se);
    for (int o = 0; o < NOUT; ++o) out[tid * NOUT + o] = sO[tid * NOUT + o] - m - ls;
  }
}

// ---------------------------------------------------------------------------
extern "C" void kernel_launch(void* const* d_in, const int* in_sizes, int n_in,
                              void* d_out, int out_size, void* d_ws, size_t ws_size,
                              hipStream_t stream) {
  (void)in_sizes; (void)n_in; (void)out_size; (void)ws_size;
  // Input leaves in setup_inputs() insertion order.
  auto F = [&](int i) { return (const float*)d_in[i]; };
  const float* pos = F(1);

  char* ws = (char*)d_ws;
  size_t off = 0;
  auto alloc = [&](size_t bytes) -> void* {
    off = (off + 255) & ~(size_t)255;
    void* p = ws + off;
    off += bytes;
    return p;
  };
  _Float16* whC1 = (_Float16*)alloc(64 * 32 * 2);
  _Float16* whC2 = (_Float16*)alloc(128 * 64 * 2);
  _Float16* whC3 = (_Float16*)alloc(1024 * 128 * 2);
  _Float16* whF1 = (_Float16*)alloc(512 * 1024 * 2);
  _Float16* whF2 = (_Float16*)alloc(256 * 512 * 2);
  _Float16* whE0 = (_Float16*)alloc(64 * 32 * 2);
  _Float16* whE1 = (_Float16*)alloc(64 * 64 * 2);
  _Float16* whE2 = (_Float16*)alloc(64 * 64 * 2);
  _Float16* whQ  = (_Float16*)alloc(64 * 64 * 2);
  _Float16* whK  = (_Float16*)alloc(64 * 64 * 2);
  _Float16* whV  = (_Float16*)alloc(64 * 64 * 2);
  _Float16* whO  = (_Float16*)alloc(64 * 64 * 2);
  _Float16* whL1 = (_Float16*)alloc(1024 * 64 * 2);
  _Float16* whH1 = (_Float16*)alloc(512 * 1024 * 2);
  _Float16* whH2 = (_Float16*)alloc(256 * 512 * 2);
  float*    stnfeat = (float*)alloc(16 * 1024 * 4);
  float*    trans   = (float*)alloc(16 * 9 * 4);
  float*    pos_t   = (float*)alloc((size_t)NPTS * 3 * 4);
  unsigned* mm      = (unsigned*)alloc(8 * 4);
  int*      nbr     = (int*)alloc((size_t)NPTS * NKNN * 4);
  _Float16* x1      = (_Float16*)alloc((size_t)NPTS * 64 * 2);
  float*    hfeat   = (float*)alloc(16 * 1024 * 4);
  _Float16* bufA = (_Float16*)alloc((size_t)NROWS * 64 * 2);
  _Float16* bufB = (_Float16*)alloc((size_t)NROWS * 64 * 2);
  _Float16* bufC = (_Float16*)alloc((size_t)NROWS * 64 * 2);
  _Float16* bufD = (_Float16*)alloc((size_t)NROWS * 64 * 2);

  auto conv = [&](_Float16* dst, const float* src, int cout, int cin, int cinp) {
    int tot = cout * cinp;
    k_convert<<<(tot + 255) / 256, 256, 0, stream>>>(dst, src, cout, cin, cinp);
  };
  conv(whC1, F(3), 64, 3, 32);
  conv(whC2, F(9), 128, 64, 64);
  conv(whC3, F(15), 1024, 128, 128);
  conv(whF1, F(21), 512, 1024, 1024);
  conv(whF2, F(27), 256, 512, 512);
  conv(whE0, F(35), 64, 6, 32);
  conv(whE1, F(41), 64, 64, 64);
  conv(whE2, F(47), 64, 64, 64);
  conv(whQ,  F(53), 64, 64, 64);
  conv(whK,  F(55), 64, 64, 64);
  conv(whV,  F(57), 64, 64, 64);
  conv(whO,  F(59), 64, 64, 64);
  conv(whL1, F(61), 1024, 64, 64);
  conv(whH1, F(67), 512, 1024, 1024);
  conv(whH2, F(73), 256, 512, 512);

  // STN
  k_stn_point<<<NB, 256, 0, stream>>>(pos,
      whC1, F(4), F(5), F(6), F(7), F(8),
      whC2, F(10), F(11), F(12), F(13), F(14),
      whC3, F(16), F(17), F(18), F(19), F(20), stnfeat);
  k_stn_fc<<<1, 256, 0, stream>>>(stnfeat,
      whF1, F(22), F(23), F(24), F(25), F(26),
      whF2, F(28), F(29), F(30), F(31), F(32),
      F(33), F(34), trans);
  k_transform<<<NPTS / 256, 256, 0, stream>>>(pos, trans, pos_t);

  // KNN normalization + neighbor search
  k_mm_init<<<1, 32, 0, stream>>>(mm);
  k_minmax<<<64, 256, 0, stream>>>(pos_t, mm);
  k_knn<<<NB * NF, 256, 0, stream>>>(pos_t, mm, nbr);

  // Edge conv MLP stack
  k_edge0<<<NROWS / 32, 256, 0, stream>>>(pos_t, nbr, whE0, F(36), F(37), F(38), F(39), F(40), bufA);
  k_gemm64<1><<<NROWS / 32, 256, 0, stream>>>(bufA, whE1, F(42), F(43), F(44), F(45), F(46), bufB);
  k_gemm64<1><<<NROWS / 32, 256, 0, stream>>>(bufB, whE2, F(48), F(49), F(50), F(51), F(52), bufA);
  // MHA
  k_gemm64<0><<<NROWS / 32, 256, 0, stream>>>(bufA, whQ, F(54), nullptr, nullptr, nullptr, nullptr, bufB);
  k_gemm64<0><<<NROWS / 32, 256, 0, stream>>>(bufA, whK, F(56), nullptr, nullptr, nullptr, nullptr, bufC);
  k_gemm64<0><<<NROWS / 32, 256, 0, stream>>>(bufA, whV, F(58), nullptr, nullptr, nullptr, nullptr, bufD);
  k_attn<<<(NPTS * 8) / 256, 256, 0, stream>>>(bufB, bufC, bufD, bufA);
  k_oproj_max<<<NROWS / 32, 256, 0, stream>>>(bufA, whO, F(60), x1);

  // lin1 + global max, classifier head
  k_lin1max<<<NB, 256, 0, stream>>>(x1, whL1, F(62), F(63), F(64), F(65), F(66), hfeat);
  k_head<<<1, 256, 0, stream>>>(hfeat,
      whH1, F(68), F(69), F(70), F(71), F(72),
      whH2, F(74), F(75), F(76), F(77), F(78),
      F(79), F(80), (float*)d_out);
}
